// SingleBlock_43404939493741
// MI455X (gfx1250) — matrix operand order repaired and checked by reading, not compiled, not run on hardware
//
#include <hip/hip_runtime.h>
#include <hip/hip_bf16.h>

// ---------------------------------------------------------------------------
// CDNA5 (gfx1250): all matmuls via v_wmma_f32_16x16x32_bf16 (wave32, fp32 acc).
// Flag-specialized template kernels so relu/acc/sigmoid/gate cost nothing when
// unused (runtime flags previously forced per-element v_cndmask selects).
// ---------------------------------------------------------------------------

typedef __bf16 v16bf __attribute__((ext_vector_type(16)));
typedef __bf16 v8bf  __attribute__((ext_vector_type(8)));
typedef __bf16 v2bf  __attribute__((ext_vector_type(2)));
typedef float  v8f   __attribute__((ext_vector_type(8)));

enum { F_RELU = 1, F_ACC = 2, F_SIG = 4 };

static __device__ __forceinline__ v8f wmma_bf16(v16bf a, v16bf b, v8f c) {
  return __builtin_amdgcn_wmma_f32_16x16x32_bf16(false, a, false, b, (short)0, c,
                                                 false, false);
}

// A-fragment (16x32 bf16): elements 0..15 = K base..base+15 (base=(lane>>4)*16).
static __device__ __forceinline__ v16bf frag_a(const __bf16* lo, const __bf16* hi) {
  v8bf l = *(const v8bf*)lo;
  v8bf h = *(const v8bf*)hi;
  return __builtin_shufflevector(l, h, 0, 1, 2, 3, 4, 5, 6, 7,
                                 8, 9, 10, 11, 12, 13, 14, 15);
}

static __device__ __forceinline__ v8f zero8() {
  v8f z;
#pragma unroll
  for (int i = 0; i < 8; ++i) z[i] = 0.f;
  return z;
}

// ---------------------------------------------------------------------------
// GEMM: C[M,N] = epilogue( act(A[M,K]) @ W^T )   W pre-transposed bf16 [N][ldw]
// FLAGS (compile-time): 1 relu on A, 2 accumulate into C, 4 sigmoid output.
// Tile BM=128, BN=128, BK=32; 256 threads = 8 waves (4x2), wave = 32x64.
// ---------------------------------------------------------------------------
template <int FLAGS>
__global__ __launch_bounds__(256) void gemm_bf16_wmma(
    const float* __restrict__ A, int lda,
    const __bf16* __restrict__ Wt, int ldw,      // [N][ldw] bf16, row stride ldw
    const float* __restrict__ bias,
    const float* __restrict__ rowmask,
    float* __restrict__ C, int ldc,
    int M, int N, int K) {
  constexpr bool kRelu = FLAGS & F_RELU;
  constexpr bool kAcc  = FLAGS & F_ACC;
  constexpr bool kSig  = FLAGS & F_SIG;

  __shared__ __bf16 sA[128 * 32];   // [row][k]
  __shared__ __bf16 sB[128 * 32];   // [col][k]

  const int tid  = threadIdx.x;
  const int m0   = blockIdx.y * 128;
  const int n0   = blockIdx.x * 128;
  const int wave = tid >> 5;
  const int lane = tid & 31;
  const int wm   = wave >> 1;       // 0..3 -> 32-row slab
  const int wn   = wave & 1;        // 0..1 -> 64-col slab

  v8f acc[2][4];
#pragma unroll
  for (int i = 0; i < 2; ++i)
#pragma unroll
    for (int j = 0; j < 4; ++j) acc[i][j] = zero8();

  // A staging: thread covers 16 consecutive floats of one row
  const int arow  = tid >> 1;            // 0..127
  const int acol  = (tid & 1) * 16;      // 0 / 16
  const int grow  = m0 + arow;
  const float keep = (grow < M) ? 1.f : 0.f;   // predication for partial M
  const float* ap = A + (size_t)(grow < M ? grow : (M - 1)) * lda + acol;
  // B staging: thread covers 16 consecutive bf16 of one Wt row
  const int bcol  = tid >> 1;            // 0..127
  const int bk0   = (tid & 1) * 16;
  const __bf16* wp = Wt + (size_t)(n0 + bcol) * ldw + bk0;

  for (int k0 = 0; k0 < K; k0 += 32) {
    // ---- stage A (fp32 -> bf16, compile-time relu, predicated rows) ----
    {
      const float4* ap4 = (const float4*)(ap + k0);
      float vbuf[16];
#pragma unroll
      for (int t = 0; t < 4; ++t) {
        float4 f = ap4[t];
        vbuf[4 * t + 0] = f.x; vbuf[4 * t + 1] = f.y;
        vbuf[4 * t + 2] = f.z; vbuf[4 * t + 3] = f.w;
      }
      v8bf lo, hi;
#pragma unroll
      for (int j = 0; j < 8; ++j) {
        float x = vbuf[j];
        if constexpr (kRelu) x = fmaxf(x, 0.f);
        lo[j] = (__bf16)(x * keep);
        float y = vbuf[8 + j];
        if constexpr (kRelu) y = fmaxf(y, 0.f);
        hi[j] = (__bf16)(y * keep);
      }
      *(v8bf*)&sA[arow * 32 + acol]     = lo;
      *(v8bf*)&sA[arow * 32 + acol + 8] = hi;
      if (k0 + 32 < K) __builtin_prefetch(ap + k0 + 32, 0, 1);
    }
    // ---- stage B: contiguous bf16 copies from pre-transposed weights ----
    {
      const v8bf* w8 = (const v8bf*)(wp + k0);
      *(v8bf*)&sB[bcol * 32 + bk0]     = w8[0];
      *(v8bf*)&sB[bcol * 32 + bk0 + 8] = w8[1];
    }
    __syncthreads();

    const int kb  = (lane >> 4) * 16;
    const int kbb = (lane >> 4) * 16;
    v16bf bfr[4];
#pragma unroll
    for (int nt = 0; nt < 4; ++nt) {
      const int col = wn * 64 + nt * 16 + (lane & 15);
      bfr[nt] = *(const v16bf*)&sB[col * 32 + kbb];
    }
#pragma unroll
    for (int mt = 0; mt < 2; ++mt) {
      const int row = wm * 32 + mt * 16 + (lane & 15);
      v16bf afr = frag_a(&sA[row * 32 + kb], &sA[row * 32 + 8 + kb]);
#pragma unroll
      for (int nt = 0; nt < 4; ++nt)
        acc[mt][nt] = wmma_bf16(afr, bfr[nt], acc[mt][nt]);
    }
    __syncthreads();
  }

  // ---- epilogue ----
#pragma unroll
  for (int mt = 0; mt < 2; ++mt) {
#pragma unroll
    for (int nt = 0; nt < 4; ++nt) {
      const int col = n0 + wn * 64 + nt * 16 + (lane & 15);
#pragma unroll
      for (int r = 0; r < 8; ++r) {
        const int row = m0 + wm * 32 + mt * 16 + ((lane >> 4) * 8) + r;
        if (row < M) {
          float v = acc[mt][nt][r];
          if (bias)              v += bias[col];
          if constexpr (kAcc)    v += C[(size_t)row * ldc + col];
          if constexpr (kSig)    v = 1.f / (1.f + __expf(-v));
          if (rowmask)           v *= rowmask[row];
          C[(size_t)row * ldc + col] = v;
        }
      }
    }
  }
}

// ---------------------------------------------------------------------------
// Fused attention, one (b, h, 32-query tile). dh=128, H=8, D=1024.
// scores = Q.K^T; logits = where(mask, s, -1e9)/sqrt(128); softmax; out = P.V.
// GATE (compile-time): (1+gate) channel scaling folded into Q/K staging.
// Dynamic LDS (81920 B): sQ[32][128] sK[64][128] bf16, sS[32][256] f32,
//                        sP[32][256] bf16, sVt[128][32] bf16.
// ---------------------------------------------------------------------------
template <bool GATE>
__global__ __launch_bounds__(256) void attn_wmma(
    const float* __restrict__ Qp, int ldq,
    const float* __restrict__ Kp, int ldk,
    const float* __restrict__ Vp, int ldv,
    const float* __restrict__ kvmask,   // [B, Lk]
    const float* __restrict__ gate,     // [B, 1024] (used iff GATE)
    float* __restrict__ Out, int ldo,
    int Lq, int Lk) {
  const int b    = blockIdx.z;
  const int h    = blockIdx.y;
  const int qt   = blockIdx.x;
  const int tid  = threadIdx.x;
  const int wave = tid >> 5;
  const int lane = tid & 31;

  extern __shared__ char smem[];
  __bf16* sQ  = (__bf16*)smem;                       // 32*128
  __bf16* sK  = sQ + 32 * 128;                       // 64*128
  float*  sS  = (float*)(sK + 64 * 128);             // 32*256
  __bf16* sP  = (__bf16*)(sS + 32 * 256);            // 32*256
  __bf16* sVt = sP + 32 * 256;                       // 128*32

  // ---- stage Q tile (gated), vectorized ----
  {
    const int r  = tid >> 3;
    const int c0 = (tid & 7) * 16;
    const float4* qp4 =
        (const float4*)(Qp + (size_t)(b * Lq + qt * 32 + r) * ldq + h * 128 + c0);
    float vbuf[16];
#pragma unroll
    for (int t = 0; t < 4; ++t) {
      float4 f = qp4[t];
      vbuf[4 * t] = f.x; vbuf[4 * t + 1] = f.y; vbuf[4 * t + 2] = f.z; vbuf[4 * t + 3] = f.w;
    }
    v8bf lo, hi;
#pragma unroll
    for (int j = 0; j < 8; ++j) {
      float x = vbuf[j], y = vbuf[8 + j];
      if constexpr (GATE) {
        x *= 1.f + gate[b * 1024 + h * 128 + c0 + j];
        y *= 1.f + gate[b * 1024 + h * 128 + c0 + 8 + j];
      }
      lo[j] = (__bf16)x;
      hi[j] = (__bf16)y;
    }
    *(v8bf*)&sQ[r * 128 + c0]     = lo;
    *(v8bf*)&sQ[r * 128 + c0 + 8] = hi;
  }

  // ---- phase 1: scores, 64 keys per stage ----
  const int mt1 = wave & 1;
  const int nt1 = wave >> 1;
  for (int kt = 0; kt < Lk; kt += 64) {
    __syncthreads();
    {
      const int kr  = tid >> 2;          // 0..63
      const int kc0 = (tid & 3) * 32;
      const float4* kp4 =
          (const float4*)(Kp + (size_t)(b * Lk + kt + kr) * ldk + h * 128 + kc0);
#pragma unroll
      for (int seg = 0; seg < 4; ++seg) {  // 4 x 8 floats
        float4 f0 = kp4[2 * seg], f1 = kp4[2 * seg + 1];
        float vb[8] = {f0.x, f0.y, f0.z, f0.w, f1.x, f1.y, f1.z, f1.w};
        v8bf o;
#pragma unroll
        for (int j = 0; j < 8; ++j) {
          float x = vb[j];
          if constexpr (GATE)
            x *= 1.f + gate[b * 1024 + h * 128 + kc0 + seg * 8 + j];
          o[j] = (__bf16)x;
        }
        *(v8bf*)&sK[kr * 128 + kc0 + seg * 8] = o;
      }
    }
    __syncthreads();

    v8f s = zero8();
    const int kb  = (lane >> 4) * 16;
    const int kbb = (lane >> 4) * 16;
#pragma unroll
    for (int ks = 0; ks < 4; ++ks) {
      const int row = mt1 * 16 + (lane & 15);
      v16bf afr = frag_a(&sQ[row * 128 + ks * 32 + kb],
                         &sQ[row * 128 + ks * 32 + kb + 8]);
      const int key = nt1 * 16 + (lane & 15);
      v16bf bfr = *(const v16bf*)&sK[key * 128 + ks * 32 + kbb];
      s = wmma_bf16(afr, bfr, s);
    }
#pragma unroll
    for (int r = 0; r < 8; ++r)
      sS[(mt1 * 16 + (lane >> 4) * 8 + r) * 256 + kt + nt1 * 16 + (lane & 15)] = s[r];
  }
  __syncthreads();

  // ---- softmax: 8 threads per row, shuffle reductions ----
  {
    const int row = tid >> 3;
    const int sub = tid & 7;
    const float rs = 0.08838834764831845f;   // 1/sqrt(128)
    float mx = -3.4e38f;
    for (int j = sub; j < Lk; j += 8) {
      float sc = sS[row * 256 + j];
      sc = (kvmask[b * Lk + j] > 0.f) ? sc : -1e9f;
      sc *= rs;
      sS[row * 256 + j] = sc;
      mx = fmaxf(mx, sc);
    }
#pragma unroll
    for (int d = 4; d > 0; d >>= 1) mx = fmaxf(mx, __shfl_xor(mx, d, 8));
    float sum = 0.f;
    for (int j = sub; j < Lk; j += 8) {
      float e = __expf(sS[row * 256 + j] - mx);
      sS[row * 256 + j] = e;
      sum += e;
    }
#pragma unroll
    for (int d = 4; d > 0; d >>= 1) sum += __shfl_xor(sum, d, 8);
    const float inv = 1.f / sum;
    for (int j = sub; j < Lk; j += 8)
      sP[row * 256 + j] = (__bf16)(sS[row * 256 + j] * inv);
  }
  __syncthreads();

  // ---- phase 3: out = P.V ----
  const int mt3   = wave & 1;
  const int npair = wave >> 1;
  v8f o0 = zero8(), o1 = zero8();
  for (int kt = 0; kt < Lk; kt += 32) {
    __syncthreads();
    {
      // stage V transposed: sVt[ch][key], packed pair-of-keys b32 stores
      const int vk2 = (tid & 15) * 2;     // even key
      const int c0  = (tid >> 4) * 8;     // 8 channels
      const float* vp0 = Vp + (size_t)(b * Lk + kt + vk2) * ldv + h * 128 + c0;
      const float* vp1 = vp0 + ldv;
      const float4* a4 = (const float4*)vp0;
      const float4* b4 = (const float4*)vp1;
      float4 fa0 = a4[0], fa1 = a4[1], fb0 = b4[0], fb1 = b4[1];
      float va[8] = {fa0.x, fa0.y, fa0.z, fa0.w, fa1.x, fa1.y, fa1.z, fa1.w};
      float vb[8] = {fb0.x, fb0.y, fb0.z, fb0.w, fb1.x, fb1.y, fb1.z, fb1.w};
#pragma unroll
      for (int j = 0; j < 8; ++j) {
        v2bf t;
        t[0] = (__bf16)va[j];
        t[1] = (__bf16)vb[j];
        *(v2bf*)&sVt[(c0 + j) * 32 + vk2] = t;
      }
    }
    __syncthreads();

    const int kb  = (lane >> 4) * 16;
    const int kbb = (lane >> 4) * 16;
    const int row = mt3 * 16 + (lane & 15);
    v16bf afr = frag_a(&sP[row * 256 + kt + kb], &sP[row * 256 + kt + kb + 8]);
    const int ch0 = npair * 32 + (lane & 15);
    v16bf b0 = *(const v16bf*)&sVt[ch0 * 32 + kbb];
    v16bf b1 = *(const v16bf*)&sVt[(ch0 + 16) * 32 + kbb];
    o0 = wmma_bf16(afr, b0, o0);
    o1 = wmma_bf16(afr, b1, o1);
  }

#pragma unroll
  for (int t = 0; t < 2; ++t) {
    const int ch = h * 128 + npair * 32 + t * 16 + (lane & 15);
    const v8f& o = t ? o1 : o0;
#pragma unroll
    for (int r = 0; r < 8; ++r) {
      const int qrow = qt * 32 + mt3 * 16 + (lane >> 4) * 8 + r;
      Out[(size_t)(b * Lq + qrow) * ldo + ch] = o[r];
    }
  }
}

// ---------------------------------------------------------------------------
// Weight convert + transpose:  out[n][k] (bf16, [N][K]) = in[k][n] (f32 [K,N])
// ---------------------------------------------------------------------------
__global__ void convT_f32_bf16(const float* __restrict__ in, __bf16* __restrict__ out,
                               int K, int N) {
  __shared__ float tile[32][33];
  const int kb = blockIdx.y * 32, nb = blockIdx.x * 32;
  const int tx = threadIdx.x, ty = threadIdx.y;   // (32, 8)
  for (int i = ty; i < 32; i += 8)
    tile[i][tx] = in[(size_t)(kb + i) * N + nb + tx];
  __syncthreads();
  for (int i = ty; i < 32; i += 8)
    out[(size_t)(nb + i) * K + kb + tx] = (__bf16)tile[tx][i];
}

__global__ void add_f32(const float* __restrict__ a, const float* __restrict__ b,
                        float* __restrict__ c, size_t n) {
  for (size_t i = blockIdx.x * (size_t)blockDim.x + threadIdx.x; i < n;
       i += (size_t)gridDim.x * blockDim.x)
    c[i] = a[i] + b[i];
}

__global__ void masked_mean(const float* __restrict__ X, const float* __restrict__ mask,
                            float* __restrict__ out, int B_, int L, int D_) {
  const int idx = blockIdx.x * blockDim.x + threadIdx.x;
  if (idx >= B_ * D_) return;
  const int b = idx / D_, d = idx % D_;
  float s = 0.f, ms = 0.f;
  for (int l = 0; l < L; ++l) {
    const float m = mask[b * L + l];
    s += X[((size_t)(b * L + l)) * D_ + d] * m;
    ms += m;
  }
  out[(size_t)b * D_ + d] = s / ms;
}

// ---------------------------------------------------------------------------
// Host orchestration
// ---------------------------------------------------------------------------
static inline void gemm(hipStream_t s, const float* A, int lda, const __bf16* Wt,
                        int ldw, const float* bias, const float* rowmask, float* C,
                        int ldc, int M, int N, int K, int flags) {
  dim3 grid(N / 128, (M + 127) / 128);
  switch (flags) {
    case 0:
      gemm_bf16_wmma<0><<<grid, 256, 0, s>>>(A, lda, Wt, ldw, bias, rowmask, C, ldc, M, N, K);
      break;
    case F_RELU:
      gemm_bf16_wmma<F_RELU><<<grid, 256, 0, s>>>(A, lda, Wt, ldw, bias, rowmask, C, ldc, M, N, K);
      break;
    case F_ACC:
      gemm_bf16_wmma<F_ACC><<<grid, 256, 0, s>>>(A, lda, Wt, ldw, bias, rowmask, C, ldc, M, N, K);
      break;
    default:  // F_RELU | F_SIG
      gemm_bf16_wmma<F_RELU | F_SIG><<<grid, 256, 0, s>>>(A, lda, Wt, ldw, bias, rowmask, C, ldc, M, N, K);
      break;
  }
}

static inline void attn(hipStream_t s, const float* Qp, const float* Kp,
                        const float* Vp, const float* mask, const float* gate,
                        float* Out, int Lq, int Lk) {
  dim3 grid(Lq / 32, 8, 64);
  if (gate)
    attn_wmma<true><<<grid, 256, 81920, s>>>(Qp, 3072, Kp, 3072, Vp, 3072, mask, gate,
                                             Out, 1024, Lq, Lk);
  else
    attn_wmma<false><<<grid, 256, 81920, s>>>(Qp, 3072, Kp, 3072, Vp, 3072, mask, gate,
                                              Out, 1024, Lq, Lk);
}

extern "C" void kernel_launch(void* const* d_in, const int* in_sizes, int n_in,
                              void* d_out, int out_size, void* d_ws, size_t ws_size,
                              hipStream_t stream) {
  (void)in_sizes; (void)n_in; (void)out_size; (void)ws_size;
  const int B = 64, NO = 256, L = 64, D = 1024;
  const int Mv = B * NO;    // 16384
  const int Mq = B * L;     // 4096

  const float* inV   = (const float*)d_in[0];
  const float* inQ   = (const float*)d_in[1];
  const float* vmask = (const float*)d_in[2];
  const float* qmask = (const float*)d_in[3];

  const float* pW[12]; const float* pB[12];
  for (int i = 0; i < 12; ++i) { pW[i] = (const float*)d_in[4 + 2 * i];
                                 pB[i] = (const float*)d_in[5 + 2 * i]; }
  // 0:v0 1:q0 2:iv 3:iq 4:ivo 5:iqo 6:v4q 7:q4v 8:dv 9:dq 10:dvo 11:dqo
  const int wK[12] = {2048, 2048, 1024, 1024, 2048, 2048, 1024, 1024, 1024, 1024, 1024, 1024};
  const int wN[12] = {1024, 1024, 3072, 3072, 1024, 1024, 1024, 1024, 3072, 3072, 1024, 1024};

  size_t off = 0;
  auto alloc = [&](size_t nbytes) -> void* {
    void* p = (char*)d_ws + off;
    off += (nbytes + 255) & ~(size_t)255;
    return p;
  };
  float* Vb[2] = {(float*)alloc((size_t)Mv * D * 4), (float*)alloc((size_t)Mv * D * 4)};
  float* Qb[2] = {(float*)alloc((size_t)Mq * D * 4), (float*)alloc((size_t)Mq * D * 4)};
  float* Vt3   = (float*)alloc((size_t)Mv * 3 * D * 4);
  float* Qt3   = (float*)alloc((size_t)Mq * 3 * D * 4);
  float* Vupd  = (float*)alloc((size_t)Mv * D * 4);
  float* Qupd  = (float*)alloc((size_t)Mq * D * 4);
  float* meanV = (float*)alloc((size_t)B * D * 4);
  float* meanQ = (float*)alloc((size_t)B * D * 4);
  float* gV4Q  = (float*)alloc((size_t)B * D * 4);
  float* gQ4V  = (float*)alloc((size_t)B * D * 4);

  __bf16* wbf[12];
  for (int i = 0; i < 12; ++i) {
    const size_t n = (size_t)wK[i] * wN[i];
    wbf[i] = (__bf16*)alloc(n * 2);
    dim3 tg(wN[i] / 32, wK[i] / 32);
    convT_f32_bf16<<<tg, dim3(32, 8), 0, stream>>>(pW[i], wbf[i], wK[i], wN[i]);
  }

  float* outV = (float*)d_out;
  float* outQ = (float*)d_out + (size_t)Mv * D;

  float* curV = Vb[0]; float* altV = Vb[1];
  float* curQ = Qb[0]; float* altQ = Qb[1];
  // initial projections: Wt is [N][K], ldw = K
  gemm(stream, inV, 2048, wbf[0], 2048, pB[0], nullptr, curV, D, Mv, D, 2048, 0);
  gemm(stream, inQ, 2048, wbf[1], 2048, pB[1], nullptr, curQ, D, Mq, D, 2048, 0);

  for (int it = 0; it < 2; ++it) {
    // ---------------- inter ----------------
    gemm(stream, curV, D, wbf[2], D, pB[2], vmask, Vt3, 3 * D, Mv, 3 * D, D, F_RELU);
    gemm(stream, curQ, D, wbf[3], D, pB[3], qmask, Qt3, 3 * D, Mq, 3 * D, D, F_RELU);
    // v_upd = mha(vq, qk, qv, q_mask)
    attn(stream, Vt3 + D, Qt3, Qt3 + 2 * D, qmask, nullptr, Vupd, NO, L);
    // q_upd = mha(qq, vk, vv, v_mask)
    attn(stream, Qt3 + D, Vt3, Vt3 + 2 * D, vmask, nullptr, Qupd, L, NO);
    // new_v = [v, v_upd] @ ivo + b  (Wt row = output col, K-split at 1024)
    gemm(stream, curV, D, wbf[4],        2048, pB[4], nullptr, altV, D, Mv, D, D, 0);
    gemm(stream, Vupd, D, wbf[4] + 1024, 2048, nullptr, nullptr, altV, D, Mv, D, D, F_ACC);
    gemm(stream, curQ, D, wbf[5],        2048, pB[5], nullptr, altQ, D, Mq, D, D, 0);
    gemm(stream, Qupd, D, wbf[5] + 1024, 2048, nullptr, nullptr, altQ, D, Mq, D, D, F_ACC);
    { float* t = curV; curV = altV; altV = t; }
    { float* t = curQ; curQ = altQ; altQ = t; }

    // ---------------- intra ----------------
    masked_mean<<<dim3((B * D + 255) / 256), 256, 0, stream>>>(curV, vmask, meanV, B, NO, D);
    masked_mean<<<dim3((B * D + 255) / 256), 256, 0, stream>>>(curQ, qmask, meanQ, B, L, D);
    gemm(stream, meanV, D, wbf[6], D, pB[6], nullptr, gV4Q, D, B, D, D, F_RELU | F_SIG);
    gemm(stream, meanQ, D, wbf[7], D, pB[7], nullptr, gQ4V, D, B, D, D, F_RELU | F_SIG);
    gemm(stream, curV, D, wbf[8], D, pB[8], vmask, Vt3, 3 * D, Mv, 3 * D, D, F_RELU);
    gemm(stream, curQ, D, wbf[9], D, pB[9], qmask, Qt3, 3 * D, Mq, 3 * D, D, F_RELU);
    attn(stream, Vt3 + D, Vt3, Vt3 + 2 * D, vmask, gQ4V, Vupd, NO, NO);
    attn(stream, Qt3 + D, Qt3, Qt3 + 2 * D, qmask, gV4Q, Qupd, L, L);
    float* Vsum = Vt3; float* Qsum = Qt3;   // reuse (attn already consumed them)
    add_f32<<<dim3(2048), 256, 0, stream>>>(curV, Vupd, Vsum, (size_t)Mv * D);
    add_f32<<<dim3(2048), 256, 0, stream>>>(curQ, Qupd, Qsum, (size_t)Mq * D);
    float* tgtV = (it == 1) ? outV : altV;
    float* tgtQ = (it == 1) ? outQ : altQ;
    gemm(stream, Vsum, D, wbf[10], D, pB[10], nullptr, tgtV, D, Mv, D, D, 0);
    gemm(stream, Qsum, D, wbf[11], D, pB[11], nullptr, tgtQ, D, Mq, D, D, 0);
    if (it == 0) {
      { float* t = curV; curV = altV; altV = t; }
      { float* t = curQ; curQ = altQ; altQ = t; }
    }
  }
}
